// Net_Integral_70789650973459
// MI455X (gfx1250) — matrix operand order repaired and checked by reading, not compiled, not run on hardware
//
#include <hip/hip_runtime.h>

typedef __attribute__((ext_vector_type(16))) _Float16 v16h;
typedef __attribute__((ext_vector_type(8)))  _Float16 v8h;
typedef __attribute__((ext_vector_type(8)))  float    v8f;

#define NZ 512
#define NX 512
#define NB 512
#define PI_F 3.14159265358979f

// Padded dims (K must be a multiple of 32 for v_wmma_f32_16x16x32_f16)
#define D0   32   // input 6  -> 32
#define H1   64   // 40 -> 64
#define H2   96   // 80 -> 96
#define H3  160   // 160
#define H4  320   // 320
#define DOUT 16   // 1 -> 16

#define MTILE 2   // row tiles (16 rows each) per wave -> M=32 rows/wave

// ---- workspace layout: f16 weight region, then f32 region ----
#define OFF_W0   0
#define OFF_W1   (OFF_W0  + D0*H1)
#define OFF_W2   (OFF_W1  + H1*H2)
#define OFF_W3   (OFF_W2  + H2*H3)
#define OFF_W4   (OFF_W3  + H3*H4)
#define OFF_W3T  (OFF_W4  + H4*DOUT)
#define OFF_W2T  (OFF_W3T + H4*H3)
#define OFF_W1T  (OFF_W2T + H3*H2)
#define OFF_W0T  (OFF_W1T + H2*H1)
#define HALVES_TOTAL (OFF_W0T + H1*DOUT)   // 153600 halves = 300 KB

#define FB0   0
#define FB1   (FB0+H1)
#define FB2   (FB1+H2)
#define FB3   (FB2+H3)
#define FB4   (FB3+H4)
#define F_W4V (FB4+DOUT)
#define F_FW  (F_W4V+H4)
#define F_CG  (F_FW+NX)
#define F_FG  (F_CG+NB)
#define F_GG  (F_FG+NZ)

// ================= fragment helpers (ISA §7.12.2 layouts) =================

// B fragment (32x16 f16): packed in ws as [tile][lane][16 halves]; two 16B loads.
__device__ inline v16h load_b_frag(const _Float16* wt, int tile, int lane) {
  const _Float16* p = wt + tile * 512 + lane * 16;
  union { v16h v; v8h h[2]; } u;
  u.h[0] = *(const v8h*)(p);
  u.h[1] = *(const v8h*)(p + 8);
  return u.v;
}

// A fragment (16x32 f16) from a row-major LDS tile s[16][Np]:
// lane l: m = l&15, kb = (l>=16)?8:0 ; elems 0..7 = K kt*32+kb.. ; 8..15 = +16.
__device__ inline v16h load_a_frag(const _Float16* s, int Np, int kt, int lane) {
  const int m  = lane & 15;
  const int kb = (lane >> 4) * 8;
  const _Float16* p = s + m * Np + kt * 32;
  union { v16h v; v8h h[2]; } u;
  u.h[0] = *(const v8h*)(p + kb);
  u.h[1] = *(const v8h*)(p + 16 + kb);
  return u.v;
}

__device__ inline v8f wmma16(v16h a, v16h b, v8f c) {
  return __builtin_amdgcn_wmma_f32_16x16x32_f16(false, a, false, b, (short)0, c,
                                                false, false);
}

// Forward layer: C = A*W ; z = C + bias ; write sin(z) / (cscale*cos(z)) to LDS.
// C layout: lane l holds col n = l&15 ; VGPR j holds row (l>=16?8:0)+j.
template <int MT, bool SSIN, bool SCOS, bool CSCALE, class AF>
__device__ inline void fwd_layer(AF afetch, int KT, int NT,
                                 const _Float16* wt, const float* bias,
                                 _Float16* sOut, _Float16* cOut,
                                 const float* cscale, int Np, int lane) {
  const int n0 = lane & 15;
  const int mb = (lane >> 4) * 8;
  const v8f vzero = {};
  for (int nt = 0; nt < NT; ++nt) {
    v8f acc[MT];
#pragma unroll
    for (int mt = 0; mt < MT; ++mt) acc[mt] = vzero;
    for (int kt = 0; kt < KT; ++kt) {
      const v16h b = load_b_frag(wt, nt * KT + kt, lane);
#pragma unroll
      for (int mt = 0; mt < MT; ++mt) acc[mt] = wmma16(afetch(mt, kt), b, acc[mt]);
    }
    const int   col = nt * 16 + n0;
    const float bv  = bias[col];
    const float cs  = CSCALE ? cscale[col] : 1.0f;
#pragma unroll
    for (int mt = 0; mt < MT; ++mt) {
#pragma unroll
      for (int j = 0; j < 8; ++j) {
        const float z   = acc[mt][j] + bv;
        const int   idx = (mt * 16 + mb + j) * Np + col;
        if (SSIN) sOut[idx] = (_Float16)__sinf(z);
        if (SCOS) cOut[idx] = (_Float16)(cs * __cosf(z));
      }
    }
  }
}

// Backward layer: C = d*W^T ; d_prev = C .* cosbuf -> LDS
template <int MT, class AF>
__device__ inline void bwd_layer(AF afetch, int KT, int NT,
                                 const _Float16* wt, const _Float16* cosb,
                                 _Float16* dOut, int Np, int lane) {
  const int n0 = lane & 15;
  const int mb = (lane >> 4) * 8;
  const v8f vzero = {};
  for (int nt = 0; nt < NT; ++nt) {
    v8f acc[MT];
#pragma unroll
    for (int mt = 0; mt < MT; ++mt) acc[mt] = vzero;
    for (int kt = 0; kt < KT; ++kt) {
      const v16h b = load_b_frag(wt, nt * KT + kt, lane);
#pragma unroll
      for (int mt = 0; mt < MT; ++mt) acc[mt] = wmma16(afetch(mt, kt), b, acc[mt]);
    }
    const int col = nt * 16 + n0;
#pragma unroll
    for (int mt = 0; mt < MT; ++mt) {
#pragma unroll
      for (int j = 0; j < 8; ++j) {
        const int idx = (mt * 16 + mb + j) * Np + col;
        dOut[idx] = (_Float16)(acc[mt][j] * (float)cosb[idx]);
      }
    }
  }
}

__device__ inline float wave_sum(float v) {
  for (int off = 16; off > 0; off >>= 1) v += __shfl_xor(v, off, 32);
  return v;
}

// ============================ pack kernel ============================

__device__ inline float maskedW(const float* W, int fi, int fo, int k, int n,
                                int blkR, int blkC) {
  if (k >= fi || n >= fo) return 0.f;
  if (blkR && (n / blkC != k / blkR)) return 0.f;
  return W[k * fo + n];
}

__device__ inline void packW(_Float16* dst, const float* W, int fi, int fo,
                             int Kp, int Np, int blkR, int blkC, bool transposed,
                             int tid, int nth) {
  const int KT = Kp / 32;
  const int total = Kp * Np;
  for (int e = tid; e < total; e += nth) {
    const int tile = e >> 9, r = e & 511, lane = r >> 4, h = r & 15;
    const int nt = tile / KT, kt = tile - nt * KT;
    const int k = kt * 32 + ((lane >> 4) << 3) + (h & 7) + ((h >> 3) << 4);
    const int n = (nt << 4) + (lane & 15);
    const float v = transposed ? maskedW(W, fi, fo, n, k, blkR, blkC)
                               : maskedW(W, fi, fo, k, n, blkR, blkC);
    dst[e] = (_Float16)v;
  }
}

__global__ void pack_kernel(const float* W0, const float* b0v, const float* W1,
                            const float* b1v, const float* W2, const float* b2v,
                            const float* W3, const float* b3v, const float* W4,
                            const float* b4v, const float* xi, const float* xi_w,
                            const float* xb, const float* xb_w, const int* btype,
                            const int* case_index, _Float16* wh, float* wf) {
  const int tid = blockIdx.x * blockDim.x + threadIdx.x;
  const int nth = gridDim.x * blockDim.x;
  // forward weights (B-fragment layout)
  packW(wh + OFF_W0, W0,   6,  40, D0, H1,  0,  0, false, tid, nth);
  packW(wh + OFF_W1, W1,  40,  80, H1, H2,  0,  0, false, tid, nth);
  packW(wh + OFF_W2, W2,  80, 160, H2, H3, 40, 80, false, tid, nth);
  packW(wh + OFF_W3, W3, 160, 320, H3, H4, 40, 80, false, tid, nth);
  packW(wh + OFF_W4, W4, 320,   1, H4, DOUT, 0, 0, false, tid, nth);
  // transposed weights for backward
  packW(wh + OFF_W3T, W3, 160, 320, H4, H3, 40, 80, true, tid, nth);
  packW(wh + OFF_W2T, W2,  80, 160, H3, H2, 40, 80, true, tid, nth);
  packW(wh + OFF_W1T, W1,  40,  80, H2, H1,  0,  0, true, tid, nth);
  packW(wh + OFF_W0T, W0,   6,  40, H1, DOUT, 0, 0, true, tid, nth);
  // padded biases + W4 vector
  for (int i = tid; i < H1; i += nth) wf[FB0 + i] = (i < 40) ? b0v[i] : 0.f;
  for (int i = tid; i < H2; i += nth) wf[FB1 + i] = (i < 80) ? b1v[i] : 0.f;
  for (int i = tid; i < H3; i += nth) wf[FB2 + i] = b2v[i];
  for (int i = tid; i < H4; i += nth) wf[FB3 + i] = b3v[i];
  for (int i = tid; i < DOUT; i += nth) wf[FB4 + i] = (i < 1) ? b4v[i] : 0.f;
  for (int i = tid; i < H4; i += nth) wf[F_W4V + i] = W4[i];
  // quadrature coefficients
  const float cf = (float)(case_index[0] + 1);
  for (int x = tid; x < NX; x += nth) {
    const float a = xi[x * 3], b = xi[x * 3 + 1], c = xi[x * 3 + 2];
    wf[F_FW + x] = sinf(PI_F * cf * a) * sinf(PI_F * b) * sinf(PI_F * c) * xi_w[x];
  }
  for (int b = tid; b < NB; b += nth) {
    const float x0 = xb[b * 3], x1 = xb[b * 3 + 1], x2 = xb[b * 3 + 2];
    const float g = sinf(cf * (x0 + x1 + x2)) * (1.0f + 0.1f * (float)btype[b]);
    const float a = 1.0f + 0.5f * cosf(x0);
    wf[F_CG + b] = a * g * xb_w[b];
  }
  for (int z = tid; z < NZ; z += nth) { wf[F_FG + z] = 0.f; wf[F_GG + z] = 0.f; }
}

// ============================ interior kernel ============================
// 1 wave = 32 rows (one z, 32 consecutive x). Fused forward, atomic reduce.

#define WAVES_I 2
#define WH_I (32 * (H1 + H2 + H3 + H4))   // halves per wave = 20480 (40 KB)

__global__ __launch_bounds__(32 * WAVES_I) void interior_kernel(
    const float* __restrict__ xi, const float* __restrict__ zc,
    const _Float16* __restrict__ wh, const float* __restrict__ wf,
    float* __restrict__ fG) {
  extern __shared__ char smem_raw[];
  const int lane = threadIdx.x & 31;
  const int wid  = threadIdx.x >> 5;
  _Float16* S1 = (_Float16*)smem_raw + wid * WH_I;
  _Float16* S2 = S1 + 32 * H1;
  _Float16* S3 = S2 + 32 * H2;
  _Float16* S4 = S3 + 32 * H3;

  // warm weights into WGP$/L2 (global_prefetch_b8)
  for (int off = threadIdx.x * 64; off < HALVES_TOTAL; off += blockDim.x * 64)
    __builtin_prefetch(wh + off, 0, 1);

  const int tile = blockIdx.x * WAVES_I + wid;
  const int row0 = tile * (16 * MTILE);
  const int zi   = row0 >> 9;          // NX = 512
  const int x0   = row0 & (NX - 1);

  // input A fragments: K = [x0,x1,x2,z0,z1,z2, 0...]
  v16h a0[MTILE];
#pragma unroll
  for (int mt = 0; mt < MTILE; ++mt) {
    union { v16h v; _Float16 e[16]; } ua;
#pragma unroll
    for (int i = 0; i < 16; ++i) ua.e[i] = (_Float16)0.0f;
    if (lane < 16) {
      const int r = x0 + mt * 16 + lane;
      ua.e[0] = (_Float16)xi[r * 3 + 0];
      ua.e[1] = (_Float16)xi[r * 3 + 1];
      ua.e[2] = (_Float16)xi[r * 3 + 2];
      ua.e[3] = (_Float16)zc[zi * 3 + 0];
      ua.e[4] = (_Float16)zc[zi * 3 + 1];
      ua.e[5] = (_Float16)zc[zi * 3 + 2];
    }
    a0[mt] = ua.v;
  }

  auto af0 = [&](int mt, int) { return a0[mt]; };
  fwd_layer<MTILE, true, false, false>(af0, D0 / 32, H1 / 16, wh + OFF_W0,
                                       wf + FB0, S1, (_Float16*)nullptr,
                                       (const float*)nullptr, H1, lane);
  auto af1 = [&](int mt, int kt) { return load_a_frag(S1 + mt * 16 * H1, H1, kt, lane); };
  fwd_layer<MTILE, true, false, false>(af1, H1 / 32, H2 / 16, wh + OFF_W1,
                                       wf + FB1, S2, (_Float16*)nullptr,
                                       (const float*)nullptr, H2, lane);
  auto af2 = [&](int mt, int kt) { return load_a_frag(S2 + mt * 16 * H2, H2, kt, lane); };
  fwd_layer<MTILE, true, false, false>(af2, H2 / 32, H3 / 16, wh + OFF_W2,
                                       wf + FB2, S3, (_Float16*)nullptr,
                                       (const float*)nullptr, H3, lane);
  auto af3 = [&](int mt, int kt) { return load_a_frag(S3 + mt * 16 * H3, H3, kt, lane); };
  fwd_layer<MTILE, true, false, false>(af3, H3 / 32, H4 / 16, wh + OFF_W3,
                                       wf + FB3, S4, (_Float16*)nullptr,
                                       (const float*)nullptr, H4, lane);

  // output layer (N padded to 16; only col 0 valid) + reduction over x
  const v8f vzero = {};
  v8f acc[MTILE];
#pragma unroll
  for (int mt = 0; mt < MTILE; ++mt) acc[mt] = vzero;
  for (int kt = 0; kt < H4 / 32; ++kt) {
    const v16h b = load_b_frag(wh + OFF_W4, kt, lane);
#pragma unroll
    for (int mt = 0; mt < MTILE; ++mt)
      acc[mt] = wmma16(load_a_frag(S4 + mt * 16 * H4, H4, kt, lane), b, acc[mt]);
  }
  float partial = 0.f;
  if ((lane & 15) == 0) {
    const float b4 = wf[FB4];
    const float* fwv = wf + F_FW;
    const int mb = (lane >> 4) * 8;
#pragma unroll
    for (int mt = 0; mt < MTILE; ++mt)
#pragma unroll
      for (int j = 0; j < 8; ++j)
        partial += (acc[mt][j] + b4) * fwv[x0 + mt * 16 + mb + j];
  }
  partial = wave_sum(partial);
  if (lane == 0) atomicAdd(&fG[zi], partial);
}

// ============================ boundary kernel ============================
// Forward (stores sin + W4-scaled cos), then transposed-GEMM backward chain.

#define WAVES_B 2
#define WH_B (32 * (H1 + H2 + H3) + 32 * (H1 + H2 + H3 + H4))  // 30720 halves

__global__ __launch_bounds__(32 * WAVES_B) void boundary_kernel(
    const float* __restrict__ xb, const float* __restrict__ zc,
    const float* __restrict__ nrm, const _Float16* __restrict__ wh,
    const float* __restrict__ wf, float* __restrict__ gG) {
  extern __shared__ char smem_raw[];
  const int lane = threadIdx.x & 31;
  const int wid  = threadIdx.x >> 5;
  _Float16* S1 = (_Float16*)smem_raw + wid * WH_B;
  _Float16* S2 = S1 + 32 * H1;
  _Float16* S3 = S2 + 32 * H2;
  _Float16* C1 = S3 + 32 * H3;
  _Float16* C2 = C1 + 32 * H1;
  _Float16* C3 = C2 + 32 * H2;
  _Float16* C4 = C3 + 32 * H3;

  for (int off = threadIdx.x * 64; off < HALVES_TOTAL; off += blockDim.x * 64)
    __builtin_prefetch(wh + off, 0, 1);

  const int tile = blockIdx.x * WAVES_B + wid;
  const int row0 = tile * (16 * MTILE);
  const int zi   = row0 >> 9;          // NB = 512
  const int b0   = row0 & (NB - 1);

  v16h a0[MTILE];
#pragma unroll
  for (int mt = 0; mt < MTILE; ++mt) {
    union { v16h v; _Float16 e[16]; } ua;
#pragma unroll
    for (int i = 0; i < 16; ++i) ua.e[i] = (_Float16)0.0f;
    if (lane < 16) {
      const int r = b0 + mt * 16 + lane;
      ua.e[0] = (_Float16)xb[r * 3 + 0];
      ua.e[1] = (_Float16)xb[r * 3 + 1];
      ua.e[2] = (_Float16)xb[r * 3 + 2];
      ua.e[3] = (_Float16)zc[zi * 3 + 0];
      ua.e[4] = (_Float16)zc[zi * 3 + 1];
      ua.e[5] = (_Float16)zc[zi * 3 + 2];
    }
    a0[mt] = ua.v;
  }

  // ---- forward: sin for next layer, cos for backward.
  //      Layer 3 stores W4[col]*cos(z4) so backward d4 = A-frag load. ----
  auto af0 = [&](int mt, int) { return a0[mt]; };
  fwd_layer<MTILE, true, true, false>(af0, D0 / 32, H1 / 16, wh + OFF_W0,
                                      wf + FB0, S1, C1, (const float*)nullptr,
                                      H1, lane);
  auto af1 = [&](int mt, int kt) { return load_a_frag(S1 + mt * 16 * H1, H1, kt, lane); };
  fwd_layer<MTILE, true, true, false>(af1, H1 / 32, H2 / 16, wh + OFF_W1,
                                      wf + FB1, S2, C2, (const float*)nullptr,
                                      H2, lane);
  auto af2 = [&](int mt, int kt) { return load_a_frag(S2 + mt * 16 * H2, H2, kt, lane); };
  fwd_layer<MTILE, true, true, false>(af2, H2 / 32, H3 / 16, wh + OFF_W2,
                                      wf + FB2, S3, C3, (const float*)nullptr,
                                      H3, lane);
  auto af3 = [&](int mt, int kt) { return load_a_frag(S3 + mt * 16 * H3, H3, kt, lane); };
  fwd_layer<MTILE, false, true, true>(af3, H3 / 32, H4 / 16, wh + OFF_W3,
                                      wf + FB3, (_Float16*)nullptr, C4,
                                      wf + F_W4V, H4, lane);

  // ---- backward chain: d4(=C4) -> d3 -> d2 -> d1 -> dX ----
  auto afd4 = [&](int mt, int kt) { return load_a_frag(C4 + mt * 16 * H4, H4, kt, lane); };
  bwd_layer<MTILE>(afd4, H4 / 32, H3 / 16, wh + OFF_W3T, C3, S3, H3, lane);  // d3 -> S3
  auto afd3 = [&](int mt, int kt) { return load_a_frag(S3 + mt * 16 * H3, H3, kt, lane); };
  bwd_layer<MTILE>(afd3, H3 / 32, H2 / 16, wh + OFF_W2T, C2, S2, H2, lane);  // d2 -> S2
  auto afd2 = [&](int mt, int kt) { return load_a_frag(S2 + mt * 16 * H2, H2, kt, lane); };
  bwd_layer<MTILE>(afd2, H2 / 32, H1 / 16, wh + OFF_W1T, C1, S1, H1, lane);  // d1 -> S1

  // dX = d1 @ W0^T ; cols 0..2 -> (grad_x G . n) * (a*g*w) ; reduce over b
  const v8f vzero = {};
  v8f acc[MTILE];
#pragma unroll
  for (int mt = 0; mt < MTILE; ++mt) acc[mt] = vzero;
  for (int kt = 0; kt < H1 / 32; ++kt) {
    const v16h b = load_b_frag(wh + OFF_W0T, kt, lane);
#pragma unroll
    for (int mt = 0; mt < MTILE; ++mt)
      acc[mt] = wmma16(load_a_frag(S1 + mt * 16 * H1, H1, kt, lane), b, acc[mt]);
  }
  const int col = lane & 15;
  const int mb  = (lane >> 4) * 8;
  float partial = 0.f;
  if (col < 3) {
    const float* cg = wf + F_CG;
#pragma unroll
    for (int mt = 0; mt < MTILE; ++mt)
#pragma unroll
      for (int j = 0; j < 8; ++j) {
        const int bi = b0 + mt * 16 + mb + j;
        partial += acc[mt][j] * nrm[bi * 3 + col] * cg[bi];
      }
  }
  partial = wave_sum(partial);
  if (lane == 0) atomicAdd(&gG[zi], partial);
}

// ============================ combine ============================

__global__ void combine_kernel(const float* __restrict__ wf, float* __restrict__ out) {
  const int z = blockIdx.x * blockDim.x + threadIdx.x;
  if (z < NZ) out[z] = wf[F_FG + z] - wf[F_GG + z];
}

// ============================ launch ============================

extern "C" void kernel_launch(void* const* d_in, const int* in_sizes, int n_in,
                              void* d_out, int out_size, void* d_ws, size_t ws_size,
                              hipStream_t stream) {
  (void)in_sizes; (void)n_in; (void)out_size; (void)ws_size;
  const float* xi    = (const float*)d_in[0];
  const float* xi_w  = (const float*)d_in[1];
  const float* xb    = (const float*)d_in[2];
  const float* xb_w  = (const float*)d_in[3];
  const float* nrm   = (const float*)d_in[4];
  const float* zc    = (const float*)d_in[5];
  const float* W0 = (const float*)d_in[6];  const float* b0 = (const float*)d_in[7];
  const float* W1 = (const float*)d_in[8];  const float* b1 = (const float*)d_in[9];
  const float* W2 = (const float*)d_in[10]; const float* b2 = (const float*)d_in[11];
  const float* W3 = (const float*)d_in[12]; const float* b3 = (const float*)d_in[13];
  const float* W4 = (const float*)d_in[14]; const float* b4 = (const float*)d_in[15];
  const int* btype = (const int*)d_in[16];
  const int* cidx  = (const int*)d_in[17];

  _Float16* wh = (_Float16*)d_ws;
  float*    wf = (float*)((char*)d_ws + (size_t)HALVES_TOTAL * sizeof(_Float16));
  float*    out = (float*)d_out;

  pack_kernel<<<96, 256, 0, stream>>>(W0, b0, W1, b1, W2, b2, W3, b3, W4, b4,
                                      xi, xi_w, xb, xb_w, btype, cidx, wh, wf);

  const int tilesI = (NZ * NX) / (16 * MTILE);
  interior_kernel<<<tilesI / WAVES_I, 32 * WAVES_I,
                    WAVES_I * WH_I * sizeof(_Float16), stream>>>(
      xi, zc, wh, wf, wf + F_FG);

  const int tilesB = (NZ * NB) / (16 * MTILE);
  boundary_kernel<<<tilesB / WAVES_B, 32 * WAVES_B,
                    WAVES_B * WH_B * sizeof(_Float16), stream>>>(
      xb, zc, nrm, wh, wf, wf + F_GG);

  combine_kernel<<<(NZ + 255) / 256, 256, 0, stream>>>(wf, out);
}
